// MoE_10943576670416
// MI455X (gfx1250) — matrix operand order repaired and checked by reading, not compiled, not run on hardware
//
#include <hip/hip_runtime.h>
#include <hip/hip_bf16.h>
#include <math.h>

// ---------------------------------------------------------------------------
// MoE (E=8, top-2, D=1024, T=8192) for gfx1250 using v_wmma_f32_16x16x32_bf16
// Routed (top-2 only) bf16 GEMMs with fp32 accumulation.
// ---------------------------------------------------------------------------

#define NEXP 8
#define TOPK 2
#define DM   1024          // d_model
#define DF   2048          // 2*d_model
#define TTOK 8192          // total tokens
#define TM   32            // token tile per ffn block (2 WMMA m-tiles)
#define FFN_THREADS 512    // 16 waves

typedef __attribute__((ext_vector_type(16))) __bf16 v16bf;
typedef __attribute__((ext_vector_type(8)))  __bf16 v8bf;
typedef __attribute__((ext_vector_type(8)))  float  v8f;

// LDS strides (elements) padded to kill bank conflicts on row-strided b128 loads
#define XS_STR 1032        // bf16, 2064 B/row
#define HS_STR 2056        // bf16, 4112 B/row
#define OS_STR 1028        // f32,  4112 B/row (exactly overlays Hs rows)

// ---------------------------------------------------------------------------
// Kernel 0: zero d_out and per-expert counts
// ---------------------------------------------------------------------------
__global__ void moe_init_kernel(float* __restrict__ out, int n4, int* __restrict__ counts) {
    int i = blockIdx.x * blockDim.x + threadIdx.x;
    if (i < NEXP) counts[i] = 0;
    if (i < n4) {
        float4 z = make_float4(0.f, 0.f, 0.f, 0.f);
        reinterpret_cast<float4*>(out)[i] = z;
    }
}

// ---------------------------------------------------------------------------
// Kernel 1: fp32 [E][R][C] -> bf16 [E][C][R]  (LDS-tiled transpose + convert)
// ---------------------------------------------------------------------------
__global__ void transpose_bf16_kernel(const float* __restrict__ src, __bf16* __restrict__ dst,
                                      int R, int C) {
    __shared__ float tile[32][33];
    const int e  = blockIdx.z;
    const int r0 = blockIdx.y * 32;
    const int c0 = blockIdx.x * 32;
    const float* s = src + (size_t)e * R * C;
    __bf16*      d = dst + (size_t)e * R * C;
    const int tx = threadIdx.x;           // 0..31
    const int ty = threadIdx.y;           // 0..7
#pragma unroll
    for (int i = 0; i < 4; ++i) {
        int r = r0 + ty + i * 8;
        tile[ty + i * 8][tx] = s[(size_t)r * C + c0 + tx];
    }
    __syncthreads();
#pragma unroll
    for (int i = 0; i < 4; ++i) {
        int c = c0 + ty + i * 8;
        d[(size_t)c * R + r0 + tx] = (__bf16)tile[tx][ty + i * 8];
    }
}

// ---------------------------------------------------------------------------
// Kernel 2: gating. One wave (32 lanes) per token.
// ---------------------------------------------------------------------------
__global__ void gate_kernel(const float* __restrict__ x, const float* __restrict__ gw,
                            int* __restrict__ counts, int* __restrict__ tokIdx,
                            float* __restrict__ tokW) {
    const int wv = threadIdx.x >> 5;
    const int lane = threadIdx.x & 31;
    const int t = blockIdx.x * 8 + wv;
    const float* xrow = x + (size_t)t * DM;

    float acc[NEXP];
#pragma unroll
    for (int e = 0; e < NEXP; ++e) acc[e] = 0.f;

    for (int d = lane; d < DM; d += 32) {
        float xv = xrow[d];
        const float* g = gw + d * NEXP;
#pragma unroll
        for (int e = 0; e < NEXP; ++e) acc[e] += xv * g[e];
    }
#pragma unroll
    for (int e = 0; e < NEXP; ++e)
#pragma unroll
        for (int off = 16; off >= 1; off >>= 1)
            acc[e] += __shfl_xor(acc[e], off, 32);

    if (lane == 0) {
        int i1 = 0; float s1 = acc[0];
#pragma unroll
        for (int e = 1; e < NEXP; ++e) if (acc[e] > s1) { s1 = acc[e]; i1 = e; }
        int i2 = -1; float s2 = -INFINITY;
#pragma unroll
        for (int e = 0; e < NEXP; ++e) if (e != i1 && acc[e] > s2) { s2 = acc[e]; i2 = e; }
        float z  = expf(s2 - s1);          // <= 1
        float p1 = 1.f / (1.f + z);
        float p2 = z * p1;
        int a1 = atomicAdd(&counts[i1], 1);
        tokIdx[i1 * TTOK + a1] = t;  tokW[i1 * TTOK + a1] = p1;
        int a2 = atomicAdd(&counts[i2], 1);
        tokIdx[i2 * TTOK + a2] = t;  tokW[i2 * TTOK + a2] = p2;
    }
}

// ---------------------------------------------------------------------------
// WMMA fragment helpers
// A (16x32 bf16), row-major source with given row ptr:
//   lane<16 : m=lane,    K = {k0..k0+15}
//   lane>=16: m=lane-16, K = {k0+16..k0+31}
// B (32x16 bf16) from [N][K] row-major: lane n = lane&15,
//   K contiguous 16 starting at k0 + (lane>>4)*16.
// ---------------------------------------------------------------------------
__device__ __forceinline__ v16bf load_a_frag(const __bf16* row, int k0, int ks) {
    v8bf lo = *reinterpret_cast<const v8bf*>(row + k0 + ks);
    v8bf hi = *reinterpret_cast<const v8bf*>(row + k0 + 8 + ks);
    v16bf a;
#pragma unroll
    for (int i = 0; i < 8; ++i) { a[i] = lo[i]; a[i + 8] = hi[i]; }
    return a;
}

__device__ __forceinline__ v8f wmma_bf16(v16bf a, v16bf b, v8f c) {
    return __builtin_amdgcn_wmma_f32_16x16x32_bf16(false, a, false, b, (short)0, c, false, false);
}

// ---------------------------------------------------------------------------
// Kernel 3: routed expert FFN + residual + layernorm + weighted combine.
// grid = (E, TTOK/TM), block = 512 (16 waves). Dynamic LDS = 197632 B.
// ---------------------------------------------------------------------------
__global__ void ffn_kernel(const float* __restrict__ x,
                           const __bf16* __restrict__ w1t,   // [E][DF][DM] bf16
                           const float* __restrict__ b1,     // [E][DF]
                           const __bf16* __restrict__ w2t,   // [E][DM][DF] bf16
                           const float* __restrict__ b2,     // [E][DM]
                           const float* __restrict__ ln_g,   // [E][DM]
                           const float* __restrict__ ln_b,   // [E][DM]
                           const int* __restrict__ counts,
                           const int* __restrict__ tokIdx,
                           const float* __restrict__ tokW,
                           float* __restrict__ out) {
    extern __shared__ __align__(16) unsigned char smem[];
    __bf16* Xs = reinterpret_cast<__bf16*>(smem);                       // [TM][XS_STR]
    __bf16* Hs = reinterpret_cast<__bf16*>(smem + TM * XS_STR * 2);     // [TM][HS_STR]
    float*  Os = reinterpret_cast<float*>(smem + TM * XS_STR * 2);      // [TM][OS_STR] overlays Hs

    __shared__ int   s_tok[TM];
    __shared__ float s_wt[TM];

    const int e     = blockIdx.x;
    const int cnt   = counts[e];
    const int tile0 = blockIdx.y * TM;
    if (tile0 >= cnt) return;                         // block-uniform

    const int tid  = threadIdx.x;
    const int wv   = tid >> 5;                        // 0..15
    const int lane = tid & 31;
    const int hi   = lane >> 4;                       // half-wave select
    const int ks   = hi << 4;                         // A contiguous: 0 or 16
    const int kb   = hi << 4;                         // B contiguous: 0 or 16
    const int mrow = lane & 15;
    const int ncol = lane & 15;

    if (tid < TM) {
        int slot = tile0 + tid;
        if (slot < cnt) { s_tok[tid] = tokIdx[e * TTOK + slot]; s_wt[tid] = tokW[e * TTOK + slot]; }
        else            { s_tok[tid] = 0;                       s_wt[tid] = 0.f; }
    }
    __syncthreads();

    // ---- gather X tile, convert fp32 -> bf16 into LDS ----
    for (int i = tid; i < TM * (DM / 4); i += FFN_THREADS) {
        int r  = i / (DM / 4);
        int c4 = i - r * (DM / 4);
        float4 v = reinterpret_cast<const float4*>(x + (size_t)s_tok[r] * DM)[c4];
        __bf16* p = Xs + r * XS_STR + c4 * 4;
        p[0] = (__bf16)v.x; p[1] = (__bf16)v.y; p[2] = (__bf16)v.z; p[3] = (__bf16)v.w;
    }
    __syncthreads();

    // ---- GEMM1: H[32][2048] = gelu(X @ W1 + b1); each wave owns 128 columns ----
    const __bf16* w1e = w1t + (size_t)e * DF * DM;
    const __bf16* arow0 = Xs + mrow * XS_STR;          // m-tile 0 rows
    const __bf16* arow1 = Xs + (mrow + 16) * XS_STR;   // m-tile 1 rows
    for (int nc = 0; nc < 2; ++nc) {
        const int nbase = wv * 128 + nc * 64;
        v8f acc[2][4];
#pragma unroll
        for (int mt = 0; mt < 2; ++mt)
#pragma unroll
            for (int t = 0; t < 4; ++t) acc[mt][t] = (v8f)(0.f);
#pragma unroll 2
        for (int k0 = 0; k0 < DM; k0 += 32) {
            v16bf a0 = load_a_frag(arow0, k0, ks);
            v16bf a1 = load_a_frag(arow1, k0, ks);
#pragma unroll
            for (int t = 0; t < 4; ++t) {
                int n = nbase + t * 16 + ncol;
                v16bf b = *reinterpret_cast<const v16bf*>(w1e + (size_t)n * DM + k0 + kb);
                acc[0][t] = wmma_bf16(a0, b, acc[0][t]);
                acc[1][t] = wmma_bf16(a1, b, acc[1][t]);
            }
        }
#pragma unroll
        for (int t = 0; t < 4; ++t) {
            int n = nbase + t * 16 + ncol;
            float bias = b1[e * DF + n];
#pragma unroll
            for (int mt = 0; mt < 2; ++mt)
#pragma unroll
                for (int j = 0; j < 8; ++j) {
                    int m = mt * 16 + j + (hi << 3);
                    float v = acc[mt][t][j] + bias;
                    float g = 0.5f * v * (1.f + erff(v * 0.70710678118654752f));  // exact gelu
                    Hs[m * HS_STR + n] = (__bf16)g;
                }
        }
    }
    __syncthreads();

    // ---- GEMM2: O[32][1024] = H @ W2; each wave owns 64 columns ----
    const __bf16* w2e = w2t + (size_t)e * DM * DF;
    const __bf16* hrow0 = Hs + mrow * HS_STR;
    const __bf16* hrow1 = Hs + (mrow + 16) * HS_STR;
    const int nbase2 = wv * 64;
    v8f acc2[2][4];
#pragma unroll
    for (int mt = 0; mt < 2; ++mt)
#pragma unroll
        for (int t = 0; t < 4; ++t) acc2[mt][t] = (v8f)(0.f);
#pragma unroll 2
    for (int k0 = 0; k0 < DF; k0 += 32) {
        v16bf a0 = load_a_frag(hrow0, k0, ks);
        v16bf a1 = load_a_frag(hrow1, k0, ks);
#pragma unroll
        for (int t = 0; t < 4; ++t) {
            int n = nbase2 + t * 16 + ncol;
            v16bf b = *reinterpret_cast<const v16bf*>(w2e + (size_t)n * DF + k0 + kb);
            acc2[0][t] = wmma_bf16(a0, b, acc2[0][t]);
            acc2[1][t] = wmma_bf16(a1, b, acc2[1][t]);
        }
    }
    __syncthreads();                                  // all waves done reading Hs

    // ---- write O (fp32, + b2) into LDS overlaying Hs ----
#pragma unroll
    for (int t = 0; t < 4; ++t) {
        int n = nbase2 + t * 16 + ncol;
        float bias = b2[e * DM + n];
#pragma unroll
        for (int mt = 0; mt < 2; ++mt)
#pragma unroll
            for (int j = 0; j < 8; ++j) {
                int m = mt * 16 + j + (hi << 3);
                Os[m * OS_STR + n] = acc2[mt][t][j] + bias;
            }
    }
    __syncthreads();

    // ---- residual + layernorm + weighted atomic combine; 2 rows per wave ----
#pragma unroll
    for (int rr = 0; rr < 2; ++rr) {
        const int r = wv * 2 + rr;
        const int token = s_tok[r];
        const float wt  = s_wt[r];
        const float* xrow = x + (size_t)token * DM;
        float sv[32];
        float sum = 0.f, sq = 0.f;
#pragma unroll
        for (int i = 0; i < 32; ++i) {
            int d = lane + i * 32;
            float s = xrow[d] + Os[r * OS_STR + d];
            sv[i] = s; sum += s; sq += s * s;
        }
#pragma unroll
        for (int off = 16; off >= 1; off >>= 1) {
            sum += __shfl_xor(sum, off, 32);
            sq  += __shfl_xor(sq,  off, 32);
        }
        float mean = sum * (1.f / DM);
        float var  = sq * (1.f / DM) - mean * mean;
        float rstd = rsqrtf(var + 1e-6f);
        if (wt != 0.f) {                              // wave-uniform
#pragma unroll
            for (int i = 0; i < 32; ++i) {
                int d = lane + i * 32;
                float y = (sv[i] - mean) * rstd * ln_g[e * DM + d] + ln_b[e * DM + d];
                atomicAdd(out + (size_t)token * DM + d, wt * y);
            }
        }
    }
}

// ---------------------------------------------------------------------------
// Host launch
// ---------------------------------------------------------------------------
extern "C" void kernel_launch(void* const* d_in, const int* in_sizes, int n_in,
                              void* d_out, int out_size, void* d_ws, size_t ws_size,
                              hipStream_t stream) {
    const float* x    = (const float*)d_in[0];
    const float* gw   = (const float*)d_in[1];
    const float* w1   = (const float*)d_in[2];
    const float* b1   = (const float*)d_in[3];
    const float* w2   = (const float*)d_in[4];
    const float* b2   = (const float*)d_in[5];
    const float* ln_g = (const float*)d_in[6];
    const float* ln_b = (const float*)d_in[7];
    float* out = (float*)d_out;

    // workspace layout (all offsets multiple of 256)
    unsigned char* ws = (unsigned char*)d_ws;
    int*    counts = (int*)(ws + 0);
    int*    tokIdx = (int*)(ws + 256);
    float*  tokW   = (float*)(ws + 256 + (size_t)NEXP * TTOK * 4);                // +262144
    __bf16* w1t    = (__bf16*)(ws + 524544);                                      // E*DF*DM bf16
    __bf16* w2t    = (__bf16*)(ws + 524544 + (size_t)NEXP * DF * DM * 2);         // E*DM*DF bf16
    // total ~67.6 MB of workspace used

    const int n4 = (TTOK * DM) / 4;                  // float4 count for output zeroing
    moe_init_kernel<<<(n4 + 255) / 256, 256, 0, stream>>>(out, n4, counts);

    transpose_bf16_kernel<<<dim3(DF / 32, DM / 32, NEXP), dim3(32, 8), 0, stream>>>(w1, w1t, DM, DF);
    transpose_bf16_kernel<<<dim3(DM / 32, DF / 32, NEXP), dim3(32, 8), 0, stream>>>(w2, w2t, DF, DM);

    gate_kernel<<<TTOK / 8, 256, 0, stream>>>(x, gw, counts, tokIdx, tokW);

    const size_t smem = (size_t)TM * XS_STR * 2 + (size_t)TM * HS_STR * 2;        // 197632 B
    ffn_kernel<<<dim3(NEXP, TTOK / TM), FFN_THREADS, smem, stream>>>(
        x, w1t, b1, w2t, b2, ln_g, ln_b, counts, tokIdx, tokW, out);
}
